// HUBNETBlockV1_28544352649673
// MI455X (gfx1250) — matrix operand order repaired and checked by reading, not compiled, not run on hardware
//
#include <hip/hip_runtime.h>
#include <hip/hip_bf16.h>

typedef __bf16 bf16_t;
typedef __attribute__((ext_vector_type(16))) __bf16 v16bf;
typedef __attribute__((ext_vector_type(8)))  __bf16 v8bf;
typedef __attribute__((ext_vector_type(8)))  float  v8f;

#define BATCH 32
#define TLEN  2048
#define DMODEL 512
#define SLOTS 64
#define MTOK  (BATCH * TLEN)      // 65536
#define SCALE_QK 0.044194173824159216f  // 1/sqrt(512)

// ---------------- workspace layout (bytes) ----------------
constexpr long OFF_CONCAT = 0;                          // bf16 (MTOK x 1024): [tokens | context]
constexpr long OFF_Q      = 134217728;                  // bf16 (MTOK x 512)   (q, later rq)
constexpr long OFF_V      = OFF_Q      + 67108864;      // bf16 (MTOK x 512)
constexpr long OFF_FUSED  = OFF_Q;                      // f32  (MTOK x 512) aliases Q+V after they die
constexpr long OFF_SCORES = OFF_V      + 67108864;      // f32  (B,T,S)
constexpr long OFF_BIND   = OFF_SCORES + 16777216;      // bf16 (B,T,S)
constexpr long OFF_BINDT  = OFF_BIND   + 8388608;       // bf16 (B,S,T)
constexpr long OFF_KKT    = OFF_BINDT  + 8388608;       // bf16 (D, B*S) transposed kk / rk
constexpr long OFF_RV     = OFF_KKT    + 2097152;       // bf16 (B,S,D)
constexpr long OFF_MEMF   = OFF_RV     + 2097152;       // f32  (B,S,D)
constexpr long OFF_MEMB   = OFF_MEMF   + 4194304;       // bf16 (B,S,D)
constexpr long OFF_WRF    = OFF_MEMB   + 2097152;       // f32  (B,S,D) write accum
constexpr long OFF_WRA    = OFF_WRF    + 4194304;       // bf16 (B,S,D) scaled write (GEMM A)
constexpr long OFF_W2     = OFF_WRA    + 2097152;       // bf16 (B,S,D) write@write_W
constexpr long OFF_G      = OFF_W2     + 2097152;       // f32  (B,S,D) gate logits
constexpr long OFF_SLOT   = OFF_G      + 4194304;       // f32  (B,S)
constexpr long OFF_POOL   = OFF_SLOT   + 8192;          // f32  (B,512)
constexpr long OFF_WB     = OFF_POOL   + 65536;         // bf16 weights: 8x(512*512) + (1024*512)

// ============================================================
// Batched bf16 WMMA GEMM: C = alpha * (A @ W) + bias
// A: (M x K) bf16 row-major;  W: (K x N) bf16 row-major.
// MODE 0: C f32 row-major | 1: C bf16 row-major | 2: C bf16 transposed (n*ldc+m)
// Block: 256 thr = 8 waves; tile 256(M) x 64(N); wave = 32x64 (8 accum).
// K must be a multiple of 64 (true for all call sites: 64/512/1024/2048).
// Register-double-buffered k-loop: loads for step k+1 issued before WMMAs
// of step k so vmem latency overlaps the matrix pipe.
// ============================================================
__device__ __forceinline__ v16bf ld_a(const bf16_t* __restrict__ p) {
  v8bf lo = *(const v8bf*)(p);
  v8bf hi = *(const v8bf*)(p + 16);
  v16bf f;
#pragma unroll
  for (int i = 0; i < 8; ++i) { f[i] = lo[i]; f[i + 8] = hi[i]; }
  return f;
}

__device__ __forceinline__ v16bf ld_b(const bf16_t* __restrict__ p, int ldw) {
  v16bf f;
#pragma unroll
  for (int i = 0; i < 8; ++i) { f[i] = p[(long)i * ldw]; f[i + 8] = p[(long)(16 + i) * ldw]; }
  return f;
}

__device__ __forceinline__ void mma8(v8f* acc, const v16bf& a0, const v16bf& a1,
                                     const v16bf* b) {
#pragma unroll
  for (int ct = 0; ct < 4; ++ct) {
    acc[ct] = __builtin_amdgcn_wmma_f32_16x16x32_bf16(
        false, a0, false, b[ct], (short)0, acc[ct], false, false);
    acc[ct + 4] = __builtin_amdgcn_wmma_f32_16x16x32_bf16(
        false, a1, false, b[ct], (short)0, acc[ct + 4], false, false);
  }
}

template <int MODE>
__global__ __launch_bounds__(256) void gemm_wmma_kernel(
    const bf16_t* __restrict__ A, long sA, int lda,
    const bf16_t* __restrict__ W, long sW, int ldw,
    const float* __restrict__ bias,
    void* __restrict__ Cv, long sC, int ldc,
    int M, int N, int K, float alpha) {
  const int bz   = blockIdx.z;
  const bf16_t* Ab = A + (long)bz * sA;
  const bf16_t* Wb = W + (long)bz * sW;
  const int lane  = threadIdx.x & 31;
  const int wv    = threadIdx.x >> 5;
  const int mbase = blockIdx.x * 256 + wv * 32;
  const int nbase = blockIdx.y * 64;
  int rA0 = mbase + (lane & 15);      if (rA0 >= M) rA0 = M - 1;
  int rA1 = mbase + 16 + (lane & 15); if (rA1 >= M) rA1 = M - 1;
  const int khalf = (lane >> 4) << 3;  // 0 or 8

  // moving pointers (no per-iteration 64-bit multiplies)
  const bf16_t* a0p = Ab + (long)rA0 * lda + khalf;
  const bf16_t* a1p = Ab + (long)rA1 * lda + khalf;
  const bf16_t* wp  = Wb + (long)khalf * ldw + nbase + (lane & 15);  // B frag: lane = N col, elems over K
  const long wstep32 = (long)32 * ldw;
  const long wstep64 = (long)64 * ldw;

  v8f acc[8];
#pragma unroll
  for (int i = 0; i < 8; ++i)
#pragma unroll
    for (int j = 0; j < 8; ++j) acc[i][j] = 0.f;

  v16bf a0A, a1A, bA[4], a0B, a1B, bB[4];
  // prologue: fragments for k = 0
  a0A = ld_a(a0p);
  a1A = ld_a(a1p);
#pragma unroll
  for (int ct = 0; ct < 4; ++ct) bA[ct] = ld_b(wp + ct * 16, ldw);

  const int iters = K >> 6;  // K/64
  for (int it = 0; it < iters - 1; ++it) {
    // prefetch k+32 into buffer B, then consume buffer A
    a0B = ld_a(a0p + 32);
    a1B = ld_a(a1p + 32);
    {
      const bf16_t* wm = wp + wstep32;
#pragma unroll
      for (int ct = 0; ct < 4; ++ct) bB[ct] = ld_b(wm + ct * 16, ldw);
    }
    mma8(acc, a0A, a1A, bA);
    // advance, prefetch k+64 into buffer A, consume buffer B
    a0p += 64; a1p += 64; wp += wstep64;
    a0A = ld_a(a0p);
    a1A = ld_a(a1p);
#pragma unroll
    for (int ct = 0; ct < 4; ++ct) bA[ct] = ld_b(wp + ct * 16, ldw);
    mma8(acc, a0B, a1B, bB);
  }
  // epilogue: last 64 of K
  a0B = ld_a(a0p + 32);
  a1B = ld_a(a1p + 32);
  {
    const bf16_t* wm = wp + wstep32;
#pragma unroll
    for (int ct = 0; ct < 4; ++ct) bB[ct] = ld_b(wm + ct * 16, ldw);
  }
  mma8(acc, a0A, a1A, bA);
  mma8(acc, a0B, a1B, bB);

  // ---- store ----
#pragma unroll
  for (int rt = 0; rt < 2; ++rt) {
    const int mr0 = mbase + rt * 16 + khalf;
#pragma unroll
    for (int ct = 0; ct < 4; ++ct) {
      const int n  = nbase + ct * 16 + (lane & 15);
      const float bv = bias ? bias[n] : 0.f;
      const v8f& a = acc[rt * 4 + ct];
      if (MODE == 2) {
        bf16_t* C = (bf16_t*)Cv + (long)bz * sC;
        if (mr0 + 7 < M) {
          v8bf o;
#pragma unroll
          for (int r = 0; r < 8; ++r) o[r] = (bf16_t)(alpha * a[r] + bv);
          *(v8bf*)(C + (long)n * ldc + mr0) = o;
        } else {
          for (int r = 0; r < 8; ++r) {
            int m = mr0 + r;
            if (m < M) C[(long)n * ldc + m] = (bf16_t)(alpha * a[r] + bv);
          }
        }
      } else if (MODE == 1) {
        bf16_t* C = (bf16_t*)Cv + (long)bz * sC;
#pragma unroll
        for (int r = 0; r < 8; ++r) {
          int m = mr0 + r;
          if (m < M) C[(long)m * ldc + n] = (bf16_t)(alpha * a[r] + bv);
        }
      } else {
        float* C = (float*)Cv + (long)bz * sC;
#pragma unroll
        for (int r = 0; r < 8; ++r) {
          int m = mr0 + r;
          if (m < M) C[(long)m * ldc + n] = alpha * a[r] + bv;
        }
      }
    }
  }
}

// ---------------- elementwise / reduction kernels ----------------

__global__ void wconv_kernel(const float* __restrict__ src, bf16_t* __restrict__ dst, int n) {
  int i = blockIdx.x * 256 + threadIdx.x;
  if (i < n) dst[i] = (bf16_t)src[i];
}

__global__ void zero_kernel(unsigned int* __restrict__ p, long n) {
  long i = (long)blockIdx.x * 256 + threadIdx.x;
  if (i < n) p[i] = 0u;
}

__global__ void initmem_kernel(const float* __restrict__ mi, float* __restrict__ memf,
                               bf16_t* __restrict__ memb) {
  long i = (long)blockIdx.x * 256 + threadIdx.x;  // over B*S*D
  float v = mi[i & (SLOTS * DMODEL - 1)];
  memf[i] = v;
  memb[i] = (bf16_t)v;
}

// token_emb gather + pos_emb + LayerNorm -> bf16 into left half of concat
__global__ __launch_bounds__(256) void embed_ln_kernel(
    const int* __restrict__ ids, const float* __restrict__ temb,
    const float* __restrict__ pemb, const float* __restrict__ g,
    const float* __restrict__ bta, bf16_t* __restrict__ concat_out) {
  const int row = blockIdx.x;
  const int t   = row & (TLEN - 1);
  const int id  = ids[row];
  const int tid = threadIdx.x;
  float x0 = temb[(long)id * DMODEL + tid]       + pemb[(long)t * DMODEL + tid];
  float x1 = temb[(long)id * DMODEL + tid + 256] + pemb[(long)t * DMODEL + tid + 256];
  __shared__ float red[256];
  red[tid] = x0 + x1;
  __syncthreads();
  for (int o = 128; o > 0; o >>= 1) { if (tid < o) red[tid] += red[tid + o]; __syncthreads(); }
  float mu = red[0] / (float)DMODEL;
  __syncthreads();
  float d0 = x0 - mu, d1 = x1 - mu;
  red[tid] = d0 * d0 + d1 * d1;
  __syncthreads();
  for (int o = 128; o > 0; o >>= 1) { if (tid < o) red[tid] += red[tid + o]; __syncthreads(); }
  float rstd = rsqrtf(red[0] / (float)DMODEL + 1e-5f);
  bf16_t* out = concat_out + (long)row * 1024;
  out[tid]       = (bf16_t)(d0 * rstd * g[tid] + bta[tid]);
  out[tid + 256] = (bf16_t)(d1 * rstd * g[tid + 256] + bta[tid + 256]);
}

// per-token: clip, top-8 of 64, softmax, mask, scatter into bind (B,T,S) and bindT (B,S,T)
__global__ __launch_bounds__(256) void topk_kernel(
    const float* __restrict__ scores, const int* __restrict__ mask,
    bf16_t* __restrict__ bind, bf16_t* __restrict__ bindT) {
  long idx = (long)blockIdx.x * 256 + threadIdx.x;
  if (idx >= (long)MTOK) return;
  const int b = (int)(idx >> 11);
  const int t = (int)(idx & (TLEN - 1));
  float s[SLOTS];
  const float* sp = scores + idx * SLOTS;
#pragma unroll
  for (int i = 0; i < SLOTS; ++i) s[i] = fminf(fmaxf(sp[i], -10.f), 10.f);
  float vals[8];
  int   idxs[8];
#pragma unroll
  for (int k = 0; k < 8; ++k) {
    float best = s[0]; int bi = 0;
    for (int i = 1; i < SLOTS; ++i) { if (s[i] > best) { best = s[i]; bi = i; } }
    vals[k] = best; idxs[k] = bi; s[bi] = -1e30f;
  }
  const float mx = vals[0];
  float e[8], sum = 0.f;
#pragma unroll
  for (int k = 0; k < 8; ++k) { e[k] = __expf(vals[k] - mx); sum += e[k]; }
  const float mf = (mask[idx] != 0) ? 1.f : 0.f;
  const float inv = mf / sum;  // masked tokens -> all zeros
  bf16_t* bp  = bind + idx * SLOTS;
  bf16_t* btp = bindT + (long)b * SLOTS * TLEN + t;
#pragma unroll
  for (int k = 0; k < 8; ++k) {
    bf16_t w = (bf16_t)(e[k] * inv);
    bp[idxs[k]] = w;
    btp[(long)idxs[k] * TLEN] = w;
  }
}

// slot_usage[b,s] = max(sum_t bindT[b,s,t], 1e-6)
__global__ __launch_bounds__(256) void slot_kernel(const bf16_t* __restrict__ bindT,
                                                   float* __restrict__ slot) {
  const int bs = blockIdx.x;
  const bf16_t* p = bindT + (long)bs * TLEN;
  float a = 0.f;
  for (int i = threadIdx.x; i < TLEN; i += 256) a += (float)p[i];
  __shared__ float red[256];
  red[threadIdx.x] = a;
  __syncthreads();
  for (int o = 128; o > 0; o >>= 1) { if (threadIdx.x < o) red[threadIdx.x] += red[threadIdx.x + o]; __syncthreads(); }
  if (threadIdx.x == 0) slot[bs] = fmaxf(red[0], 1e-6f);
}

__global__ void scale_write_kernel(const float* __restrict__ wf, const float* __restrict__ slot,
                                   bf16_t* __restrict__ out) {
  long i = (long)blockIdx.x * 256 + threadIdx.x;  // B*S*D
  out[i] = (bf16_t)(wf[i] / slot[i >> 9]);
}

// memory = LN(0.9*memory + 0.1*sigmoid(g)*w2) ; writes f32 + bf16 copies
__global__ __launch_bounds__(256) void update_kernel(
    const bf16_t* __restrict__ w2, const float* __restrict__ g,
    float* __restrict__ memf, bf16_t* __restrict__ memb,
    const float* __restrict__ mg, const float* __restrict__ mb) {
  const int r = blockIdx.x;  // B*S rows
  const int tid = threadIdx.x;
  float x[2];
#pragma unroll
  for (int j = 0; j < 2; ++j) {
    int d = tid + j * 256;
    float w2v = (float)w2[(long)r * DMODEL + d];
    float gv  = g[(long)r * DMODEL + d];
    float sig = 1.f / (1.f + __expf(-gv));
    x[j] = 0.9f * memf[(long)r * DMODEL + d] + 0.1f * sig * w2v;
  }
  __shared__ float red[256];
  red[tid] = x[0] + x[1];
  __syncthreads();
  for (int o = 128; o > 0; o >>= 1) { if (tid < o) red[tid] += red[tid + o]; __syncthreads(); }
  float mu = red[0] / (float)DMODEL;
  __syncthreads();
  float d0 = x[0] - mu, d1 = x[1] - mu;
  red[tid] = d0 * d0 + d1 * d1;
  __syncthreads();
  for (int o = 128; o > 0; o >>= 1) { if (tid < o) red[tid] += red[tid + o]; __syncthreads(); }
  float rstd = rsqrtf(red[0] / (float)DMODEL + 1e-5f);
  float y0 = d0 * rstd * mg[tid] + mb[tid];
  float y1 = d1 * rstd * mg[tid + 256] + mb[tid + 256];
  memf[(long)r * DMODEL + tid]       = y0;
  memf[(long)r * DMODEL + tid + 256] = y1;
  memb[(long)r * DMODEL + tid]       = (bf16_t)y0;
  memb[(long)r * DMODEL + tid + 256] = (bf16_t)y1;
}

// masked mean-pool over T
__global__ __launch_bounds__(256) void pool_kernel(const float* __restrict__ fused,
                                                   const int* __restrict__ mask,
                                                   float* __restrict__ pooled) {
  const int b = blockIdx.x >> 1;
  const int d = (blockIdx.x & 1) * 256 + threadIdx.x;
  __shared__ float ml[TLEN];
  __shared__ float red[256];
  float mpart = 0.f;
  for (int i = threadIdx.x; i < TLEN; i += 256) {
    float m = (mask[(long)b * TLEN + i] != 0) ? 1.f : 0.f;
    ml[i] = m;
    mpart += m;
  }
  red[threadIdx.x] = mpart;
  __syncthreads();
  for (int o = 128; o > 0; o >>= 1) { if (threadIdx.x < o) red[threadIdx.x] += red[threadIdx.x + o]; __syncthreads(); }
  const float msum = fmaxf(red[0], 1.f);
  float acc = 0.f;
  const float* fp = fused + (long)b * TLEN * DMODEL + d;
  for (int t = 0; t < TLEN; ++t) acc += fp[(long)t * DMODEL] * ml[t];
  pooled[(long)b * DMODEL + d] = acc / msum;
}

__global__ void cls_kernel(const float* __restrict__ pooled, const float* __restrict__ W,
                           const float* __restrict__ bias, float* __restrict__ out) {
  int tid = threadIdx.x;
  if (tid >= BATCH * 8) return;
  int b = tid >> 3, c = tid & 7;
  float a = bias[c];
  for (int d = 0; d < DMODEL; ++d) a += pooled[(long)b * DMODEL + d] * W[d * 8 + c];
  out[tid] = a;
}

// ---------------- host-side launcher ----------------
static inline void launch_gemm(int mode, const bf16_t* A, long sA, int lda,
                               const bf16_t* W, long sW, int ldw, const float* bias,
                               void* C, long sC, int ldc, int M, int N, int K,
                               float alpha, int batch, hipStream_t st) {
  dim3 grid((M + 255) / 256, N / 64, batch), blk(256);
  if (mode == 0)      gemm_wmma_kernel<0><<<grid, blk, 0, st>>>(A, sA, lda, W, sW, ldw, bias, C, sC, ldc, M, N, K, alpha);
  else if (mode == 1) gemm_wmma_kernel<1><<<grid, blk, 0, st>>>(A, sA, lda, W, sW, ldw, bias, C, sC, ldc, M, N, K, alpha);
  else                gemm_wmma_kernel<2><<<grid, blk, 0, st>>>(A, sA, lda, W, sW, ldw, bias, C, sC, ldc, M, N, K, alpha);
}

extern "C" void kernel_launch(void* const* d_in, const int* in_sizes, int n_in,
                              void* d_out, int out_size, void* d_ws, size_t ws_size,
                              hipStream_t stream) {
  (void)in_sizes; (void)n_in; (void)out_size; (void)ws_size;
  const int*   ids    = (const int*)d_in[0];
  const int*   amask  = (const int*)d_in[1];
  const float* temb   = (const float*)d_in[2];
  const float* pemb   = (const float*)d_in[3];
  const float* minit  = (const float*)d_in[4];
  const float* qW  = (const float*)d_in[5],  *qb  = (const float*)d_in[6];
  const float* kW  = (const float*)d_in[7],  *kb  = (const float*)d_in[8];
  const float* vW  = (const float*)d_in[9],  *vb  = (const float*)d_in[10];
  const float* wW  = (const float*)d_in[11], *wb  = (const float*)d_in[12];
  const float* gW  = (const float*)d_in[13], *gb  = (const float*)d_in[14];
  const float* rqW = (const float*)d_in[15], *rqb = (const float*)d_in[16];
  const float* rkW = (const float*)d_in[17], *rkb = (const float*)d_in[18];
  const float* rvW = (const float*)d_in[19], *rvb = (const float*)d_in[20];
  const float* fW  = (const float*)d_in[21], *fb  = (const float*)d_in[22];
  const float* cW  = (const float*)d_in[23], *cb  = (const float*)d_in[24];
  const float* tlg = (const float*)d_in[25], *tlb = (const float*)d_in[26];
  const float* mlg = (const float*)d_in[27], *mlb = (const float*)d_in[28];

  char* ws = (char*)d_ws;
  bf16_t* concat = (bf16_t*)(ws + OFF_CONCAT);
  bf16_t* qbuf   = (bf16_t*)(ws + OFF_Q);
  bf16_t* vbuf   = (bf16_t*)(ws + OFF_V);
  float*  fused  = (float*)(ws + OFF_FUSED);
  float*  scores = (float*)(ws + OFF_SCORES);
  bf16_t* bind   = (bf16_t*)(ws + OFF_BIND);
  bf16_t* bindT  = (bf16_t*)(ws + OFF_BINDT);
  bf16_t* kkT    = (bf16_t*)(ws + OFF_KKT);
  bf16_t* rvbuf  = (bf16_t*)(ws + OFF_RV);
  float*  memf   = (float*)(ws + OFF_MEMF);
  bf16_t* memb   = (bf16_t*)(ws + OFF_MEMB);
  float*  wrf    = (float*)(ws + OFF_WRF);
  bf16_t* wra    = (bf16_t*)(ws + OFF_WRA);
  bf16_t* w2b    = (bf16_t*)(ws + OFF_W2);
  float*  gbuf   = (float*)(ws + OFF_G);
  float*  slot   = (float*)(ws + OFF_SLOT);
  float*  pooled = (float*)(ws + OFF_POOL);
  bf16_t* wgt    = (bf16_t*)(ws + OFF_WB);
  const int WSZ = DMODEL * DMODEL;  // 262144
  bf16_t *qWb = wgt, *kWb = wgt + WSZ, *vWb = wgt + 2 * WSZ, *wWb = wgt + 3 * WSZ,
         *gWb = wgt + 4 * WSZ, *rqWb = wgt + 5 * WSZ, *rkWb = wgt + 6 * WSZ,
         *rvWb = wgt + 7 * WSZ, *fWb = wgt + 8 * WSZ;  // fuse: 1024*512

  // --- weight conversion (f32 -> bf16, layout preserved K x N) ---
  const float* srcs[9] = {qW, kW, vW, wW, gW, rqW, rkW, rvW, fW};
  bf16_t* dsts[9] = {qWb, kWb, vWb, wWb, gWb, rqWb, rkWb, rvWb, fWb};
  for (int i = 0; i < 9; ++i) {
    int n = (i == 8) ? 2 * WSZ : WSZ;
    wconv_kernel<<<(n + 255) / 256, 256, 0, stream>>>(srcs[i], dsts[i], n);
  }

  // --- embed + LN -> concat left half ; memory init ---
  embed_ln_kernel<<<MTOK, 256, 0, stream>>>(ids, temb, pemb, tlg, tlb, concat);
  initmem_kernel<<<(BATCH * SLOTS * DMODEL) / 256, 256, 0, stream>>>(minit, memf, memb);

  // --- loop-invariant GEMMs: q = tokens@q_W, v = tokens@v_W ---
  launch_gemm(1, concat, 0, 1024, qWb, 0, DMODEL, qb, qbuf, 0, DMODEL, MTOK, DMODEL, DMODEL, 1.f, 1, stream);
  launch_gemm(1, concat, 0, 1024, vWb, 0, DMODEL, vb, vbuf, 0, DMODEL, MTOK, DMODEL, DMODEL, 1.f, 1, stream);

  const long zn = (2 * 8388608) / 4;  // zero bind + bindT (contiguous) as u32
  for (int step = 0; step < 4; ++step) {
    // kk = memory@k_W + k_b, stored transposed (D x B*S) to serve as B operand of scores
    launch_gemm(2, memb, 0, DMODEL, kWb, 0, DMODEL, kb, kkT, 0, BATCH * SLOTS,
                BATCH * SLOTS, DMODEL, DMODEL, 1.f, 1, stream);
    // scores[b] = q[b] @ kk[b]^T * scale   (batched, N=64)
    launch_gemm(0, qbuf, (long)TLEN * DMODEL, DMODEL, kkT, SLOTS, BATCH * SLOTS, nullptr,
                scores, (long)TLEN * SLOTS, SLOTS, TLEN, SLOTS, DMODEL, SCALE_QK, BATCH, stream);
    zero_kernel<<<(unsigned)((zn + 255) / 256), 256, 0, stream>>>((unsigned int*)bind, zn);
    topk_kernel<<<MTOK / 256, 256, 0, stream>>>(scores, amask, bind, bindT);
    slot_kernel<<<BATCH * SLOTS, 256, 0, stream>>>(bindT, slot);
    // write[b] = bindT[b] @ v[b]   (M=64, K=2048)
    launch_gemm(0, bindT, (long)SLOTS * TLEN, TLEN, vbuf, (long)TLEN * DMODEL, DMODEL, nullptr,
                wrf, (long)SLOTS * DMODEL, DMODEL, SLOTS, DMODEL, TLEN, 1.f, BATCH, stream);
    scale_write_kernel<<<(BATCH * SLOTS * DMODEL) / 256, 256, 0, stream>>>(wrf, slot, wra);
    // write2 = write@write_W + write_b (bf16 out, feeds gate GEMM)
    launch_gemm(1, wra, 0, DMODEL, wWb, 0, DMODEL, wb, w2b, 0, DMODEL,
                BATCH * SLOTS, DMODEL, DMODEL, 1.f, 1, stream);
    // g = write2@gate_W + gate_b
    launch_gemm(0, w2b, 0, DMODEL, gWb, 0, DMODEL, gb, gbuf, 0, DMODEL,
                BATCH * SLOTS, DMODEL, DMODEL, 1.f, 1, stream);
    update_kernel<<<BATCH * SLOTS, 256, 0, stream>>>(w2b, gbuf, memf, memb, mlg, mlb);
  }

  // --- read phase ---
  launch_gemm(1, concat, 0, 1024, rqWb, 0, DMODEL, rqb, qbuf, 0, DMODEL, MTOK, DMODEL, DMODEL, 1.f, 1, stream);
  launch_gemm(2, memb, 0, DMODEL, rkWb, 0, DMODEL, rkb, kkT, 0, BATCH * SLOTS,
              BATCH * SLOTS, DMODEL, DMODEL, 1.f, 1, stream);
  launch_gemm(1, memb, 0, DMODEL, rvWb, 0, DMODEL, rvb, rvbuf, 0, DMODEL,
              BATCH * SLOTS, DMODEL, DMODEL, 1.f, 1, stream);
  launch_gemm(0, qbuf, (long)TLEN * DMODEL, DMODEL, kkT, SLOTS, BATCH * SLOTS, nullptr,
              scores, (long)TLEN * SLOTS, SLOTS, TLEN, SLOTS, DMODEL, SCALE_QK, BATCH, stream);
  zero_kernel<<<(unsigned)((zn + 255) / 256), 256, 0, stream>>>((unsigned int*)bind, zn);
  topk_kernel<<<MTOK / 256, 256, 0, stream>>>(scores, amask, bind, bindT);
  // context[b] = bind[b] @ rv[b], bf16 directly into right half of concat
  launch_gemm(1, bind, (long)TLEN * SLOTS, SLOTS, rvbuf, (long)SLOTS * DMODEL, DMODEL, nullptr,
              concat + DMODEL, (long)TLEN * 1024, 1024, TLEN, DMODEL, SLOTS, 1.f, BATCH, stream);
  // fused = concat @ fuse_W + fuse_b (K=1024)
  launch_gemm(0, concat, 0, 1024, fWb, 0, DMODEL, fb, fused, 0, DMODEL,
              MTOK, DMODEL, 1024, 1.f, 1, stream);
  pool_kernel<<<BATCH * 2, 256, 0, stream>>>(fused, amask, pooled);
  cls_kernel<<<1, 256, 0, stream>>>(pooled, cW, cb, (float*)d_out);
}
